// GraphAttentionEmbedding_90967407329294
// MI455X (gfx1250) — matrix operand order repaired and checked by reading, not compiled, not run on hardware
//
#include <hip/hip_runtime.h>
#include <hip/hip_bf16.h>
#include <math.h>

#define NN 50000
#define EE 250000
#define DD 256
#define TDD 64
#define EDD 320
#define HH 4
#define CC 64

typedef __attribute__((ext_vector_type(16))) __bf16 v16bf;
typedef __attribute__((ext_vector_type(8)))  __bf16 v8bf;
typedef __attribute__((ext_vector_type(8)))  float  v8f;

// ---------- scalar helpers ----------
__device__ __forceinline__ unsigned short f2bf(float a) {  // RNE
  union { float f; unsigned u; } x; x.f = a;
  return (unsigned short)((x.u + 0x7FFFu + ((x.u >> 16) & 1u)) >> 16);
}
__device__ __forceinline__ unsigned pack2_bf16(float a, float b) {
  return (unsigned)f2bf(a) | ((unsigned)f2bf(b) << 16);
}
__device__ __forceinline__ float bflo(unsigned u) {
  union { unsigned u; float f; } x; x.u = u << 16; return x.f;
}
__device__ __forceinline__ float bfhi(unsigned u) {
  union { unsigned u; float f; } x; x.u = u & 0xFFFF0000u; return x.f;
}
// monotonic float->uint key so atomicMax(uint) == float max
__device__ __forceinline__ unsigned fkey(float f) {
  union { float f; unsigned u; } x; x.f = f;
  return (x.u & 0x80000000u) ? ~x.u : (x.u | 0x80000000u);
}
__device__ __forceinline__ float funkey(unsigned k) {
  union { unsigned u; float f; } x;
  x.u = (k & 0x80000000u) ? (k & 0x7FFFFFFFu) : ~k;
  return x.f;
}

// ---------- WMMA fragment loads (16x16x32 bf16; contiguous k order used for both A and B) ----------
// A (16x32, MxK): lane L = row m=L%16; k in [off,off+16),
// off=(L>=16)*16  ->  two contiguous 16-byte loads from a bf16 row.
__device__ __forceinline__ v16bf load_a_bf16(const unsigned short* rowk0, int lane) {
  union { v16bf v; v8bf h[2]; } a;
  const int off = (lane >> 4) * 16;
  a.h[0] = *(const v8bf*)(rowk0 + off);
  a.h[1] = *(const v8bf*)(rowk0 + off + 8);
  return a.v;
}
__device__ __forceinline__ v8f wmma_bf16(v16bf a, v16bf b, v8f c) {
  return __builtin_amdgcn_wmma_f32_16x16x32_bf16(false, a, false, b, (short)0, c,
                                                 false, false);
}

// ---------- prepass kernels (bandwidth bound; conversion VALU is free) ----------
__global__ void k_cvt_weights(const float* __restrict__ Wq, const float* __restrict__ Wk,
                              const float* __restrict__ Wv, const float* __restrict__ Wskip,
                              const float* __restrict__ We,
                              unsigned short* __restrict__ wnode,
                              unsigned short* __restrict__ we_b) {
  int i = blockIdx.x * blockDim.x + threadIdx.x;
  const int nnode = 4 * 256 * 256;
  if (i < nnode) {
    int mat = i >> 16, off = i & 0xFFFF;
    const float* src = (mat == 0) ? Wq : (mat == 1) ? Wk : (mat == 2) ? Wv : Wskip;
    wnode[i] = f2bf(src[off]);
  } else {
    int j = i - nnode;
    if (j < 256 * EDD) we_b[j] = f2bf(We[j]);
  }
}

__global__ void k_f32_to_bf16_pairs(const float* __restrict__ src,
                                    unsigned* __restrict__ dst, int npairs) {
  int i = blockIdx.x * blockDim.x + threadIdx.x;
  if (i >= npairs) return;
  dst[i] = pack2_bf16(src[2 * i], src[2 * i + 1]);
}

// time encoding straight to bf16: teb[e][c] = bf16(cos(rel_t*w[c]+b[c]))
__global__ void k_time_enc(const float* __restrict__ last_update, const float* __restrict__ t,
                           const int* __restrict__ ei, const float* __restrict__ tw,
                           const float* __restrict__ tb, unsigned* __restrict__ teb) {
  int i = blockIdx.x * blockDim.x + threadIdx.x;  // pair index
  if (i >= EE * TDD / 2) return;
  int e = i >> 5, c2 = (i & 31) * 2;
  float rel = last_update[ei[e]] - t[e];
  float c0 = cosf(rel * tw[c2] + tb[c2]);
  float c1 = cosf(rel * tw[c2 + 1] + tb[c2 + 1]);
  teb[i] = pack2_bf16(c0, c1);
}

__global__ void k_init(unsigned* __restrict__ mkey, float* __restrict__ denom) {
  int i = blockIdx.x * blockDim.x + threadIdx.x;
  if (i < NN * HH) { mkey[i] = 0u; denom[i] = 0.0f; }
}

// ---------- GEMM kernels ----------
// Weights staged in LDS once per workgroup (CDNA5: up to 320KB/WGP), then
// amortized across 25 row tiles. Block = 128 threads = 4 waves:
// strip = (tid>>5)&1 (128-column half), rg = tid>>6 (row-tile parity).
// B fragments come from LDS (ds_load, DScnt) so WMMAs don't stall on global
// loadcnt; the only global stream in the K-loop is the A rows.

#define RT_NODE 25  // row tiles per workgroup; 3125 = 125 * 25
__global__ void k_node_gemm(const unsigned short* __restrict__ xb,
                            const unsigned short* __restrict__ wnode,
                            const float* __restrict__ bq, const float* __restrict__ bk,
                            const float* __restrict__ bv, const float* __restrict__ bskip,
                            unsigned short* __restrict__ q, unsigned short* __restrict__ kk,
                            unsigned short* __restrict__ v, float* __restrict__ out) {
  extern __shared__ unsigned char smem_raw[];
  unsigned short* Wl = (unsigned short*)smem_raw;  // 256*256 bf16 = 128 KB

  const int tid = threadIdx.x;
  const int lane = tid & 31;
  const int strip = (tid >> 5) & 1;
  const int rg = tid >> 6;
  const int mat = blockIdx.y;
  const unsigned short* W = wnode + (size_t)mat * 256 * 256;
  const float* bias = (mat == 0) ? bq : (mat == 1) ? bk : (mat == 2) ? bv : bskip;

  // stage full weight matrix into LDS (8192 x 16B)
  {
    const uint4* gsrc = (const uint4*)W;
    uint4* ldst = (uint4*)Wl;
    for (int i = tid; i < 8192; i += 128) ldst[i] = gsrc[i];
  }
  __syncthreads();

  const int n = lane & 15;
  const int koff = (lane >> 4) * 16;
  // B (32x16 KxN): lane holds col n, 16 contiguous K -> 32B from LDS per tile
  const unsigned short* bbase = Wl + (size_t)(strip * 128 + n) * DD + koff;

  for (int rt = rg; rt < RT_NODE; rt += 2) {
    const int m0 = blockIdx.x * (RT_NODE * 16) + rt * 16;
    const unsigned short* arow = xb + (size_t)(m0 + n) * DD;  // row m = lane&15
    v8f acc[8] = {};
    for (int k0 = 0; k0 < DD; k0 += 32) {
      v16bf a = load_a_bf16(arow + k0, lane);
      const unsigned short* bp = bbase + k0;
#pragma unroll
      for (int j = 0; j < 8; ++j)
        acc[j] = wmma_bf16(a, *(const v16bf*)(bp + (size_t)j * 16 * DD), acc[j]);
    }
    // C/D: VGPR r -> row m=(lane>=16)*8+r, col = lane&15
    const int mr0 = (lane >> 4) * 8;
    if (mat < 3) {
      unsigned short* dst = (mat == 0) ? q : (mat == 1) ? kk : v;
#pragma unroll
      for (int j = 0; j < 8; ++j) {
        int col = strip * 128 + j * 16 + n;
        float bb = bias[col];
#pragma unroll
        for (int r = 0; r < 8; ++r)
          dst[(size_t)(m0 + mr0 + r) * 256 + col] = f2bf(acc[j][r] + bb);
      }
    } else {
#pragma unroll
      for (int j = 0; j < 8; ++j) {
        int col = strip * 128 + j * 16 + n;
        float bb = bias[col];
#pragma unroll
        for (int r = 0; r < 8; ++r)
          out[(size_t)(m0 + mr0 + r) * 256 + col] = acc[j][r] + bb;
      }
    }
  }
}

// e = [te | msg] @ We^T, bf16 out. We (320x256 bf16 = 160 KB) staged in LDS.
#define RT_EDGE 25  // 15625 = 625 * 25
__global__ void k_edge_gemm(const unsigned short* __restrict__ teb,
                            const unsigned short* __restrict__ msgb,
                            const unsigned short* __restrict__ we_b,
                            unsigned short* __restrict__ eb) {
  extern __shared__ unsigned char smem_raw[];
  unsigned short* Wl = (unsigned short*)smem_raw;  // 256*320 bf16 = 160 KB

  const int tid = threadIdx.x;
  const int lane = tid & 31;
  const int strip = (tid >> 5) & 1;
  const int rg = tid >> 6;

  {
    const uint4* gsrc = (const uint4*)we_b;
    uint4* ldst = (uint4*)Wl;
    for (int i = tid; i < 10240; i += 128) ldst[i] = gsrc[i];
  }
  __syncthreads();

  const int n = lane & 15;
  const int koff = (lane >> 4) * 16;
  const unsigned short* bbase = Wl + (size_t)(strip * 128 + n) * EDD + koff;

  for (int rt = rg; rt < RT_EDGE; rt += 2) {
    const int m0 = blockIdx.x * (RT_EDGE * 16) + rt * 16;
    const unsigned short* terow = teb + (size_t)(m0 + n) * TDD;
    const unsigned short* mrow = msgb + (size_t)(m0 + n) * 256;
    v8f acc[8] = {};
#pragma unroll
    for (int s = 0; s < 2; ++s) {  // K = 0..63 from time encoding
      v16bf a = load_a_bf16(terow + s * 32, lane);
      const unsigned short* bp = bbase + s * 32;
#pragma unroll
      for (int j = 0; j < 8; ++j)
        acc[j] = wmma_bf16(a, *(const v16bf*)(bp + (size_t)j * 16 * EDD), acc[j]);
    }
    for (int s = 0; s < 8; ++s) {  // K = 64..319 from msg
      v16bf a = load_a_bf16(mrow + s * 32, lane);
      const unsigned short* bp = bbase + 64 + s * 32;
#pragma unroll
      for (int j = 0; j < 8; ++j)
        acc[j] = wmma_bf16(a, *(const v16bf*)(bp + (size_t)j * 16 * EDD), acc[j]);
    }
    const int mr0 = (lane >> 4) * 8;
#pragma unroll
    for (int j = 0; j < 8; ++j)
#pragma unroll
      for (int r = 0; r < 8; ++r)
        eb[(size_t)(m0 + mr0 + r) * 256 + strip * 128 + j * 16 + n] = f2bf(acc[j][r]);
  }
}

// ---------- segment softmax + scatter (atomic passes) ----------
__global__ void k_alpha(const unsigned short* __restrict__ q, const unsigned short* __restrict__ k,
                        const unsigned short* __restrict__ eb, const int* __restrict__ ei,
                        float* __restrict__ alraw, unsigned* __restrict__ mkey) {
  int idx = blockIdx.x * blockDim.x + threadIdx.x;
  if (idx >= EE * HH) return;
  int e = idx >> 2, h = idx & 3;
  int src = ei[e], dst = ei[EE + e];
  const uint4* q4 = (const uint4*)(q + (size_t)dst * 256 + h * CC);
  const uint4* k4 = (const uint4*)(k + (size_t)src * 256 + h * CC);
  const uint4* e4 = (const uint4*)(eb + (size_t)e * 256 + h * CC);
  float a = 0.f;
#pragma unroll
  for (int i = 0; i < 8; ++i) {
    uint4 qa = q4[i], ka = k4[i], ea = e4[i];
    a += bflo(qa.x) * (bflo(ka.x) + bflo(ea.x));
    a += bfhi(qa.x) * (bfhi(ka.x) + bfhi(ea.x));
    a += bflo(qa.y) * (bflo(ka.y) + bflo(ea.y));
    a += bfhi(qa.y) * (bfhi(ka.y) + bfhi(ea.y));
    a += bflo(qa.z) * (bflo(ka.z) + bflo(ea.z));
    a += bfhi(qa.z) * (bfhi(ka.z) + bfhi(ea.z));
    a += bflo(qa.w) * (bflo(ka.w) + bflo(ea.w));
    a += bfhi(qa.w) * (bfhi(ka.w) + bfhi(ea.w));
  }
  a *= 0.125f;  // 1/sqrt(64)
  alraw[idx] = a;
  atomicMax(&mkey[(size_t)dst * HH + h], fkey(a));
}

__global__ void k_expsum(const int* __restrict__ ei, const unsigned* __restrict__ mkey,
                         float* __restrict__ alraw, float* __restrict__ denom) {
  int idx = blockIdx.x * blockDim.x + threadIdx.x;
  if (idx >= EE * HH) return;
  int e = idx >> 2, h = idx & 3;
  int dst = ei[EE + e];
  float mm = funkey(mkey[(size_t)dst * HH + h]);
  float ex = expf(alraw[idx] - mm);
  alraw[idx] = ex;
  atomicAdd(&denom[(size_t)dst * HH + h], ex);
}

__global__ void k_scatter(const unsigned short* __restrict__ v, const unsigned short* __restrict__ eb,
                          const int* __restrict__ ei, const float* __restrict__ alraw,
                          const float* __restrict__ denom, float* __restrict__ out) {
  int idx = blockIdx.x * blockDim.x + threadIdx.x;
  if (idx >= EE * HH) return;
  int e = idx >> 2, h = idx & 3;
  int src = ei[e], dst = ei[EE + e];
  float a = alraw[idx] / (denom[(size_t)dst * HH + h] + 1e-16f);
  const uint4* v4 = (const uint4*)(v + (size_t)src * 256 + h * CC);
  const uint4* e4 = (const uint4*)(eb + (size_t)e * 256 + h * CC);
  float* op = out + (size_t)dst * 256 + h * CC;
#pragma unroll
  for (int i = 0; i < 8; ++i) {
    uint4 va = v4[i], ea = e4[i];
    atomicAdd(&op[8 * i + 0], (bflo(va.x) + bflo(ea.x)) * a);
    atomicAdd(&op[8 * i + 1], (bfhi(va.x) + bfhi(ea.x)) * a);
    atomicAdd(&op[8 * i + 2], (bflo(va.y) + bflo(ea.y)) * a);
    atomicAdd(&op[8 * i + 3], (bfhi(va.y) + bfhi(ea.y)) * a);
    atomicAdd(&op[8 * i + 4], (bflo(va.z) + bflo(ea.z)) * a);
    atomicAdd(&op[8 * i + 5], (bfhi(va.z) + bfhi(ea.z)) * a);
    atomicAdd(&op[8 * i + 6], (bflo(va.w) + bflo(ea.w)) * a);
    atomicAdd(&op[8 * i + 7], (bfhi(va.w) + bfhi(ea.w)) * a);
  }
}

extern "C" void kernel_launch(void* const* d_in, const int* in_sizes, int n_in,
                              void* d_out, int out_size, void* d_ws, size_t ws_size,
                              hipStream_t stream) {
  (void)in_sizes; (void)n_in; (void)out_size; (void)ws_size;
  const float* x           = (const float*)d_in[0];
  const float* last_update = (const float*)d_in[1];
  const float* t           = (const float*)d_in[2];
  const float* msg         = (const float*)d_in[3];
  const int*   ei          = (const int*)d_in[4];
  const float* tw          = (const float*)d_in[5];
  const float* tb          = (const float*)d_in[6];
  const float* Wq          = (const float*)d_in[7];
  const float* bq          = (const float*)d_in[8];
  const float* Wk          = (const float*)d_in[9];
  const float* bk          = (const float*)d_in[10];
  const float* Wv          = (const float*)d_in[11];
  const float* bv          = (const float*)d_in[12];
  const float* We          = (const float*)d_in[13];
  const float* Wskip       = (const float*)d_in[14];
  const float* bskip       = (const float*)d_in[15];
  float* out = (float*)d_out;

  char* ws = (char*)d_ws;
  size_t off = 0;
  auto alloc = [&](size_t bytes) -> void* {
    off = (off + 255) & ~(size_t)255;
    void* p = ws + off;
    off += bytes;
    return p;
  };
  unsigned short* wnode = (unsigned short*)alloc((size_t)4 * 256 * 256 * 2);
  unsigned short* we_b  = (unsigned short*)alloc((size_t)EDD * 256 * 2);
  unsigned short* xb    = (unsigned short*)alloc((size_t)NN * 256 * 2);
  unsigned short* msgb  = (unsigned short*)alloc((size_t)EE * 256 * 2);
  unsigned short* teb   = (unsigned short*)alloc((size_t)EE * TDD * 2);
  unsigned short* q     = (unsigned short*)alloc((size_t)NN * 256 * 2);
  unsigned short* k     = (unsigned short*)alloc((size_t)NN * 256 * 2);
  unsigned short* v     = (unsigned short*)alloc((size_t)NN * 256 * 2);
  unsigned short* eb    = (unsigned short*)alloc((size_t)EE * 256 * 2);
  unsigned* mkey = (unsigned*)alloc((size_t)NN * HH * 4);
  float* denom   = (float*)alloc((size_t)NN * HH * 4);
  float* alraw   = (float*)alloc((size_t)EE * HH * 4);

  const int cvt_total = 4 * 256 * 256 + EDD * 256;
  k_cvt_weights<<<(cvt_total + 255) / 256, 256, 0, stream>>>(Wq, Wk, Wv, Wskip, We, wnode, we_b);
  const int xpairs = NN * 256 / 2;
  k_f32_to_bf16_pairs<<<(xpairs + 255) / 256, 256, 0, stream>>>(x, (unsigned*)xb, xpairs);
  const int mpairs = EE * 256 / 2;
  k_f32_to_bf16_pairs<<<(mpairs + 255) / 256, 256, 0, stream>>>(msg, (unsigned*)msgb, mpairs);
  k_time_enc<<<(EE * TDD / 2 + 255) / 256, 256, 0, stream>>>(last_update, t, ei, tw, tb,
                                                             (unsigned*)teb);
  k_init<<<(NN * HH + 255) / 256, 256, 0, stream>>>(mkey, denom);
  k_node_gemm<<<dim3(NN / 16 / RT_NODE, 4), 128, 256 * 256 * 2, stream>>>(
      xb, wnode, bq, bk, bv, bskip, q, k, v, out);
  k_edge_gemm<<<EE / 16 / RT_EDGE, 128, EDD * 256 * 2, stream>>>(teb, msgb, we_b, eb);
  k_alpha<<<(EE * HH + 255) / 256, 256, 0, stream>>>(q, k, eb, ei, alraw, mkey);
  k_expsum<<<(EE * HH + 255) / 256, 256, 0, stream>>>(ei, mkey, alraw, denom);
  k_scatter<<<(EE * HH + 255) / 256, 256, 0, stream>>>(v, eb, ei, alraw, denom, out);
}
